// SynergyPredictor_86586540687515
// MI455X (gfx1250) — matrix operand order repaired and checked by reading, not compiled, not run on hardware
//
#include <hip/hip_runtime.h>
#include <hip/hip_bf16.h>
#include <hip/hip_fp16.h>

// ---------------------------------------------------------------------------
// Problem constants (match reference)
// ---------------------------------------------------------------------------
constexpr int ND   = 100000;
constexpr int NC   = 20000;
constexpr int E_DD = 1600000;
constexpr int E_DC = 800000;
constexpr int E_CD = 800000;
constexpr int DDIM = 256;
constexpr int CDIM = 512;
constexpr int HDIM = 128;
constexpr int B_Q  = 100000;

// GEMM tiling
constexpr int BM  = 64;          // rows per block (4 waves x 16)
constexpr int BK  = 32;          // K per WMMA step (16x16x32 f16)
constexpr int LDT = BK + 8;      // padded LDS row stride (halves): 80B rows
constexpr int NT  = HDIM / 16;   // 8 N-tiles of 16

typedef __attribute__((ext_vector_type(16))) _Float16     v16h;
typedef __attribute__((ext_vector_type(8)))  float        v8f;
typedef __attribute__((ext_vector_type(4)))  unsigned int v4u;
typedef __attribute__((ext_vector_type(8)))  int          v8i;
typedef __attribute__((ext_vector_type(4)))  int          v4i;

#if defined(__gfx1250__) && __has_builtin(__builtin_amdgcn_tensor_load_to_lds) && \
    __has_builtin(__builtin_amdgcn_s_wait_tensorcnt)
#define USE_TDM 1
#else
#define USE_TDM 0
#endif

#if defined(__AMDGCN__)
#if USE_TDM
#warning "CDNA5: TDM tensor_load_to_lds W-staging path ENABLED"
#else
#warning "CDNA5: TDM builtin missing -> manual W staging fallback"
#endif
#endif

struct Term {
  const void*     A;    // activations: f32 [M,lda] (af16=0) or f16 [M,lda] (af16=1)
  const int*      idx;  // optional row-gather indices (length M)
  const float*    cnt;  // optional per-row divisor (mean aggregation)
  const _Float16* W;    // [128, ldw] weights (pre-converted f16)
  int lda, ldw, K, af16;
};

// A fragment (16x32, f16): lane L -> row M=L&15, g=L>>4.
// elems 0..7 = K g*8..g*8+7 ; elems 8..15 = K 16+g*8..16+g*8+7
__device__ __forceinline__ v16h frag_a(const _Float16* tile) {
  const int lane = threadIdx.x & 31;
  const int r = lane & 15, g = lane >> 4;
  const _Float16* p = tile + r * LDT + g * 8;
  v16h f;
  ((float4*)&f)[0] = *(const float4*)(p);
  ((float4*)&f)[1] = *(const float4*)(p + 16);
  return f;
}

// B fragment (32x16, f16): lane L -> col N=L&15, g=L>>4; elems 0..7 = K g*8..g*8+7 ; elems 8..15 = K 16+g*8..16+g*8+7
__device__ __forceinline__ v16h frag_b(const _Float16* tile) {
  const int lane = threadIdx.x & 31;
  const int n = lane & 15, g = lane >> 4;
  const _Float16* p = tile + n * LDT + g * 8;
  v16h f;
  ((float4*)&f)[0] = *(const float4*)(p);
  ((float4*)&f)[1] = *(const float4*)(p + 16);
  return f;
}

// ---------------------------------------------------------------------------
// Fused multi-term WMMA GEMM: C[M,128] (f16) = sum_t A_t[idx]@W_t^T (+b0+b1)(relu?)
// ---------------------------------------------------------------------------
__global__ __launch_bounds__(128)
void gemm_fused(Term t0, Term t1, Term t2, Term t3,
                const float* bias0, const float* bias1,
                _Float16* __restrict__ C, int M, int relu)
{
  __shared__ _Float16 As[BM * LDT];     // 64 x 40 halves (80B rows)
  __shared__ _Float16 Ws[HDIM * LDT];   // 128 x 40 halves

  const int tid  = threadIdx.x;
  const int wave = tid >> 5;
  const int lane = tid & 31;
  const int m0   = blockIdx.x * BM;

  v8f acc[NT];
  v8f vzero = {};
  #pragma unroll
  for (int i = 0; i < NT; ++i) acc[i] = vzero;

#if USE_TDM
  const unsigned lds_ws =
      (unsigned)(size_t)(__attribute__((address_space(3))) _Float16*)Ws;
#endif

  auto run_term = [&](const Term& T) {
    if (T.A == nullptr) return;
    for (int k0 = 0; k0 < T.K; k0 += BK) {
      // ---- stage A chunk (64x32) ----
      if (T.af16) {
        const _Float16* Af = (const _Float16*)T.A;
        // 4 threads per row (8 halves each), 2 passes of 32 rows
        const int r0 = tid >> 2;
        const int c  = (tid & 3) * 8;
        #pragma unroll
        for (int i = 0; i < 2; ++i) {
          const int r  = i * 32 + r0;
          const int gm = m0 + r;
          union { uint4 u; _Float16 h[8]; } pk;
          pk.u = make_uint4(0u, 0u, 0u, 0u);
          if (gm < M) {
            const int row = T.idx ? T.idx[gm] : gm;
            pk.u = *(const uint4*)&Af[(size_t)row * T.lda + (k0 + c)];
            if (T.cnt) {
              const _Float16 s = (_Float16)(1.0f / fmaxf(T.cnt[row], 1.0f));
              #pragma unroll
              for (int j = 0; j < 8; ++j) pk.h[j] *= s;
            }
          }
          *(uint4*)&As[r * LDT + c] = pk.u;   // ds_store_b128
        }
      } else {
        const float* Af = (const float*)T.A;
        // 8 threads per row (float4 each), 4 passes of 16 rows; two-phase
        const int ar = tid >> 3;
        const int ac = (tid & 7) * 4;
        float4 av[4];
        #pragma unroll
        for (int i = 0; i < 4; ++i) {
          const int gm = m0 + i * 16 + ar;
          float4 v = make_float4(0.f, 0.f, 0.f, 0.f);
          if (gm < M)
            v = *(const float4*)&Af[(size_t)gm * T.lda + (k0 + ac)];
          av[i] = v;
        }
        #pragma unroll
        for (int i = 0; i < 4; ++i) {
          union { _Float16 h[4]; uint2 u; } pk;
          pk.h[0] = (_Float16)av[i].x; pk.h[1] = (_Float16)av[i].y;
          pk.h[2] = (_Float16)av[i].z; pk.h[3] = (_Float16)av[i].w;
          *(uint2*)&As[(i * 16 + ar) * LDT + ac] = pk.u;   // ds_store_b64
        }
      }

      // ---- stage W chunk (128x32 f16) ----
#if USE_TDM
      if (tid < 32) {
        const unsigned long long ga = (unsigned long long)(size_t)(T.W + k0);
        v4u g0;
        g0[0] = 1u;                                   // count=1 valid descriptor
        g0[1] = lds_ws;                               // lds_addr (bytes)
        g0[2] = (unsigned)ga;                         // global_addr[31:0]
        g0[3] = (unsigned)((ga >> 32) & 0x1FFFFFFu)   // global_addr[56:32]
                | (2u << 30);                         // type=2 (image)
        v8i g1;
        g1[0] = (1 << 16)      // data_size = 1 -> 2 bytes
              | (1 << 20)      // pad_enable
              | (3 << 22)      // pad_interval code 3 -> every 16 DWORDs (64B row)
              | (3 << 25);     // pad_amount  code 3 -> 4 DWORDs (16B) pad
        g1[1] = (int)(32u << 16);   // tensor_dim0 = 32   (bits 79:48, low 16)
        g1[2] = (int)(128u << 16);  // dim0 hi | tensor_dim1 = 128 (bits 111:80)
        g1[3] = (int)(32u << 16);   // dim1 hi | tile_dim0 = 32    (bits 127:112)
        g1[4] = 128;                // tile_dim1 = 128 | tile_dim2 = 0
        g1[5] = T.ldw;              // tensor_dim0_stride (elements)
        g1[6] = 0;
        g1[7] = 0;
        v4i z4 = {};
#if __clang_major__ >= 23
        v8i z8 = {};
        __builtin_amdgcn_tensor_load_to_lds(g0, g1, z4, z4, z8, 0);
#else
        __builtin_amdgcn_tensor_load_to_lds(g0, g1, z4, z4, 0);
#endif
        __builtin_amdgcn_s_wait_tensorcnt((short)0);
      }
#else
      #pragma unroll
      for (int i = 0; i < 4; ++i) {
        const int e = i * 128 + tid;
        const int r = e >> 2, c = (e & 3) * 8;
        *(float4*)&Ws[r * LDT + c] =
            *(const float4*)&T.W[(size_t)r * T.ldw + (k0 + c)];
      }
#endif
      __syncthreads();

      const v16h a = frag_a(&As[(wave * 16) * LDT]);
      #pragma unroll
      for (int nt = 0; nt < NT; ++nt) {
        const v16h b = frag_b(&Ws[(nt * 16) * LDT]);
        acc[nt] = __builtin_amdgcn_wmma_f32_16x16x32_f16(
            false, a, false, b, (short)0, acc[nt], false, false);
      }
      __syncthreads();
    }
  };
  run_term(t0); run_term(t1); run_term(t2); run_term(t3);

  // epilogue: C/D layout -> lane L, vgpr v: M = v + 8*(L>>4), N = L&15
  const int mrow = wave * 16 + (lane >> 4) * 8;
  const int ncol = lane & 15;
  #pragma unroll
  for (int nt = 0; nt < NT; ++nt) {
    const int n = nt * 16 + ncol;
    const float bb = (bias0 ? bias0[n] : 0.0f) + (bias1 ? bias1[n] : 0.0f);
    #pragma unroll
    for (int v = 0; v < 8; ++v) {
      const int gm = m0 + mrow + v;
      if (gm < M) {
        float val = acc[nt][v] + bb;
        if (relu) val = fmaxf(val, 0.0f);
        C[(size_t)gm * HDIM + n] = (_Float16)val;
      }
    }
  }
}

// ---------------------------------------------------------------------------
// Weight pre-conversion f32 -> f16
// ---------------------------------------------------------------------------
__global__ __launch_bounds__(256)
void w_to_f16(const float* __restrict__ src, _Float16* __restrict__ dst, int n)
{
  const int i = blockIdx.x * 256 + threadIdx.x;
  if (i < n) dst[i] = (_Float16)src[i];
}

// ---------------------------------------------------------------------------
// Edge scatter-add in packed f16: agg[dst] += x[src]
// 64 threads = 64 __half2 pairs per 128-feature row; global_atomic_pk_add_f16
// ---------------------------------------------------------------------------
constexpr int EPB = 16;
__global__ __launch_bounds__(64)
void scatter_add(const _Float16* __restrict__ x, const int* __restrict__ src,
                 const int* __restrict__ dst, _Float16* __restrict__ agg, int ne)
{
  const int h  = threadIdx.x;            // pair index 0..63
  const int e0 = blockIdx.x * EPB;
  #pragma unroll
  for (int i = 0; i < EPB; ++i) {
    const int e = e0 + i;
    if (e >= ne) return;
    if (i + 1 < EPB && e + 1 < ne)
      __builtin_prefetch(&x[(size_t)src[e + 1] * HDIM + 2 * h], 0, 0);
    const int s = src[e], d = dst[e];
    const __half2 v = ((const __half2*)(x + (size_t)s * HDIM))[h];
    atomicAdd(((__half2*)(agg + (size_t)d * HDIM)) + h, v);
  }
}

__global__ __launch_bounds__(256)
void count_edges(const int* __restrict__ dst, float* __restrict__ cnt, int ne)
{
  const int e = blockIdx.x * 256 + threadIdx.x;
  if (e < ne) atomicAdd(&cnt[dst[e]], 1.0f);
}

// ---------------------------------------------------------------------------
// Final head: out[b] = h[b,:] . m2_w + m2_b  (one wave32 per row, f32 accum)
// ---------------------------------------------------------------------------
__global__ __launch_bounds__(256)
void head_dot(const _Float16* __restrict__ h, const float* __restrict__ w,
              const float* __restrict__ b, float* __restrict__ out, int nrows)
{
  const int row  = (blockIdx.x * 256 + threadIdx.x) >> 5;
  const int lane = threadIdx.x & 31;
  if (row >= nrows) return;
  const _Float16* hp = h + (size_t)row * HDIM;
  float s = 0.0f;
  #pragma unroll
  for (int i = 0; i < HDIM / 32; ++i)
    s += (float)hp[lane + i * 32] * w[lane + i * 32];
  #pragma unroll
  for (int off = 16; off > 0; off >>= 1) s += __shfl_xor(s, off, 32);
  if (lane == 0) out[row] = s + b[0];
}

// ---------------------------------------------------------------------------
// Orchestration
// ---------------------------------------------------------------------------
extern "C" void kernel_launch(void* const* d_in, const int* in_sizes, int n_in,
                              void* d_out, int out_size, void* d_ws, size_t ws_size,
                              hipStream_t stream)
{
  (void)in_sizes; (void)n_in; (void)out_size; (void)ws_size;

  const float* x_drug = (const float*)d_in[0];
  const float* x_cell = (const float*)d_in[1];
  const int* dd_src = (const int*)d_in[2];
  const int* dd_dst = (const int*)d_in[3];
  const int* dc_src = (const int*)d_in[4];
  const int* dc_dst = (const int*)d_in[5];
  const int* cd_src = (const int*)d_in[6];
  const int* cd_dst = (const int*)d_in[7];
  const int* q_da = (const int*)d_in[8];
  const int* q_db = (const int*)d_in[9];
  const int* q_c  = (const int*)d_in[10];
  const float* wd = (const float*)d_in[11]; const float* bd = (const float*)d_in[12];
  const float* wc = (const float*)d_in[13]; const float* bc = (const float*)d_in[14];
  const float* dd1_wl=(const float*)d_in[15]; const float* dd1_bl=(const float*)d_in[16]; const float* dd1_wr=(const float*)d_in[17];
  const float* cd1_wl=(const float*)d_in[18]; const float* cd1_bl=(const float*)d_in[19]; const float* cd1_wr=(const float*)d_in[20];
  const float* dc1_wl=(const float*)d_in[21]; const float* dc1_bl=(const float*)d_in[22]; const float* dc1_wr=(const float*)d_in[23];
  const float* dd2_wl=(const float*)d_in[24]; const float* dd2_bl=(const float*)d_in[25]; const float* dd2_wr=(const float*)d_in[26];
  const float* cd2_wl=(const float*)d_in[27]; const float* cd2_bl=(const float*)d_in[28]; const float* cd2_wr=(const float*)d_in[29];
  const float* dc2_wl=(const float*)d_in[30]; const float* dc2_bl=(const float*)d_in[31]; const float* dc2_wr=(const float*)d_in[32];
  const float* m1_w=(const float*)d_in[33]; const float* m1_b=(const float*)d_in[34];
  const float* m2_w=(const float*)d_in[35]; const float* m2_b=(const float*)d_in[36];
  float* out = (float*)d_out;

  // workspace carve-up: f16 activations + f32 counts
  _Float16* hp = (_Float16*)d_ws;
  auto take16 = [&](size_t n) { _Float16* r = hp; hp += n; return r; };
  _Float16* xd     = take16((size_t)ND * HDIM);
  _Float16* xc     = take16((size_t)NC * HDIM);
  _Float16* hd     = take16((size_t)ND * HDIM);
  _Float16* hc     = take16((size_t)NC * HDIM);
  _Float16* od     = take16((size_t)ND * HDIM);
  _Float16* oc     = take16((size_t)NC * HDIM);
  _Float16* agg_d1 = take16((size_t)ND * HDIM);   // dd -> drug
  _Float16* agg_d2 = take16((size_t)ND * HDIM);   // cd -> drug
  _Float16* agg_c  = take16((size_t)NC * HDIM);   // dc -> cell
  _Float16* hh     = take16((size_t)B_Q * HDIM);
  _Float16* wdh    = take16((size_t)HDIM * DDIM);
  _Float16* wch    = take16((size_t)HDIM * CDIM);
  _Float16* whh[12];
  for (int i = 0; i < 12; ++i) whh[i] = take16((size_t)HDIM * HDIM);
  _Float16* m1h    = take16((size_t)HDIM * 3 * HDIM);

  float* fp = (float*)hp;
  auto take32 = [&](size_t n) { float* r = fp; fp += n; return r; };
  float* cnt_dd = take32(ND);
  float* cnt_cd = take32(ND);
  float* cnt_dc = take32(NC);

  // pre-convert all weight matrices to f16
  auto cvt = [&](const float* s, _Float16* d, int n) {
    w_to_f16<<<(n + 255) / 256, 256, 0, stream>>>(s, d, n);
  };
  cvt(wd, wdh, HDIM * DDIM);
  cvt(wc, wch, HDIM * CDIM);
  const float* wsrc[12] = {dd1_wl, dd1_wr, cd1_wl, cd1_wr, dc1_wl, dc1_wr,
                           dd2_wl, dd2_wr, cd2_wl, cd2_wr, dc2_wl, dc2_wr};
  for (int i = 0; i < 12; ++i) cvt(wsrc[i], whh[i], HDIM * HDIM);
  cvt(m1_w, m1h, HDIM * 3 * HDIM);

  auto Tf = [](const void* A, const int* idx, const float* cnt, const _Float16* W,
               int lda, int ldw, int K, int af16) {
    Term t{A, idx, cnt, W, lda, ldw, K, af16}; return t;
  };
  const Term TN = {nullptr, nullptr, nullptr, nullptr, 0, 0, 0, 0};

  const int gN = (ND + BM - 1) / BM;
  const int gC = (NC + BM - 1) / BM;
  const int gB = (B_Q + BM - 1) / BM;

  // input projections (f32 inputs -> f16 activations)
  gemm_fused<<<gN, 128, 0, stream>>>(Tf(x_drug, nullptr, nullptr, wdh, DDIM, DDIM, DDIM, 0),
                                     TN, TN, TN, bd, nullptr, xd, ND, 0);
  gemm_fused<<<gC, 128, 0, stream>>>(Tf(x_cell, nullptr, nullptr, wch, CDIM, CDIM, CDIM, 0),
                                     TN, TN, TN, bc, nullptr, xc, NC, 0);

  // edge counts (shared by both layers)
  hipMemsetAsync(cnt_dd, 0, (size_t)ND * 4, stream);
  hipMemsetAsync(cnt_cd, 0, (size_t)ND * 4, stream);
  hipMemsetAsync(cnt_dc, 0, (size_t)NC * 4, stream);
  count_edges<<<(E_DD + 255) / 256, 256, 0, stream>>>(dd_dst, cnt_dd, E_DD);
  count_edges<<<(E_CD + 255) / 256, 256, 0, stream>>>(cd_dst, cnt_cd, E_CD);
  count_edges<<<(E_DC + 255) / 256, 256, 0, stream>>>(dc_dst, cnt_dc, E_DC);

  // ---- layer 1 ----
  hipMemsetAsync(agg_d1, 0, (size_t)ND * HDIM * 2, stream);
  hipMemsetAsync(agg_d2, 0, (size_t)ND * HDIM * 2, stream);
  hipMemsetAsync(agg_c,  0, (size_t)NC * HDIM * 2, stream);
  scatter_add<<<(E_DD + EPB - 1) / EPB, 64, 0, stream>>>(xd, dd_src, dd_dst, agg_d1, E_DD);
  scatter_add<<<(E_CD + EPB - 1) / EPB, 64, 0, stream>>>(xc, cd_src, cd_dst, agg_d2, E_CD);
  scatter_add<<<(E_DC + EPB - 1) / EPB, 64, 0, stream>>>(xd, dc_src, dc_dst, agg_c,  E_DC);

  gemm_fused<<<gN, 128, 0, stream>>>(
      Tf(agg_d1, nullptr, cnt_dd, whh[0], HDIM, HDIM, HDIM, 1),
      Tf(xd,     nullptr, nullptr, whh[1], HDIM, HDIM, HDIM, 1),
      Tf(agg_d2, nullptr, cnt_cd, whh[2], HDIM, HDIM, HDIM, 1),
      Tf(xd,     nullptr, nullptr, whh[3], HDIM, HDIM, HDIM, 1),
      dd1_bl, cd1_bl, hd, ND, 1);
  gemm_fused<<<gC, 128, 0, stream>>>(
      Tf(agg_c, nullptr, cnt_dc, whh[4], HDIM, HDIM, HDIM, 1),
      Tf(xc,    nullptr, nullptr, whh[5], HDIM, HDIM, HDIM, 1),
      TN, TN, dc1_bl, nullptr, hc, NC, 1);

  // ---- layer 2 ----
  hipMemsetAsync(agg_d1, 0, (size_t)ND * HDIM * 2, stream);
  hipMemsetAsync(agg_d2, 0, (size_t)ND * HDIM * 2, stream);
  hipMemsetAsync(agg_c,  0, (size_t)NC * HDIM * 2, stream);
  scatter_add<<<(E_DD + EPB - 1) / EPB, 64, 0, stream>>>(hd, dd_src, dd_dst, agg_d1, E_DD);
  scatter_add<<<(E_CD + EPB - 1) / EPB, 64, 0, stream>>>(hc, cd_src, cd_dst, agg_d2, E_CD);
  scatter_add<<<(E_DC + EPB - 1) / EPB, 64, 0, stream>>>(hd, dc_src, dc_dst, agg_c,  E_DC);

  gemm_fused<<<gN, 128, 0, stream>>>(
      Tf(agg_d1, nullptr, cnt_dd, whh[6], HDIM, HDIM, HDIM, 1),
      Tf(hd,     nullptr, nullptr, whh[7], HDIM, HDIM, HDIM, 1),
      Tf(agg_d2, nullptr, cnt_cd, whh[8], HDIM, HDIM, HDIM, 1),
      Tf(hd,     nullptr, nullptr, whh[9], HDIM, HDIM, HDIM, 1),
      dd2_bl, cd2_bl, od, ND, 0);
  gemm_fused<<<gC, 128, 0, stream>>>(
      Tf(agg_c, nullptr, cnt_dc, whh[10], HDIM, HDIM, HDIM, 1),
      Tf(hc,    nullptr, nullptr, whh[11], HDIM, HDIM, HDIM, 1),
      TN, TN, dc2_bl, nullptr, oc, NC, 0);

  // ---- MLP head: concat via gathered terms, m1_w split along K ----
  gemm_fused<<<gB, 128, 0, stream>>>(
      Tf(od, q_da, nullptr, m1h + 0 * HDIM, HDIM, 3 * HDIM, HDIM, 1),
      Tf(od, q_db, nullptr, m1h + 1 * HDIM, HDIM, 3 * HDIM, HDIM, 1),
      Tf(oc, q_c,  nullptr, m1h + 2 * HDIM, HDIM, 3 * HDIM, HDIM, 1),
      TN, m1_b, nullptr, hh, B_Q, 1);

  head_dot<<<(B_Q * 32 + 255) / 256, 256, 0, stream>>>(hh, m2_w, m2_b, out, B_Q);
}
